// TransformerBlock_40767829573918
// MI455X (gfx1250) — matrix operand order repaired and checked by reading, not compiled, not run on hardware
//
#include <hip/hip_runtime.h>

// ---------------------------------------------------------------------------
// Transformer block for MI455X (gfx1250, wave32, WMMA).
// All GEMMs (projections, MLP, attention QK^T / PV) use
// v_wmma_f32_16x16x32_bf16 with fp32 accumulation. Attention softmax row
// reductions use DPP (16-lane rows match the WMMA C-fragment layout).
// ---------------------------------------------------------------------------

#define D_MODEL  1024
#define NUM_HEADS 16
#define HEAD_DIM  64
#define D_FF     4096
#define SEQ      2048
#define BATCH    4
#define MTOT     (BATCH * SEQ)   // 8192 rows

typedef __bf16 bf16;
typedef __attribute__((ext_vector_type(16))) __bf16 v16bf;
typedef __attribute__((ext_vector_type(8)))  __bf16 v8bf;
typedef __attribute__((ext_vector_type(8)))  float  v8f;

__device__ __forceinline__ v16bf cat8(v8bf a, v8bf b) {
  return __builtin_shufflevector(a, b, 0, 1, 2, 3, 4, 5, 6, 7,
                                       8, 9, 10, 11, 12, 13, 14, 15);
}

// ---- DPP 16-lane row reductions (rows of 16 == C-fragment half-wave) ------
template <int CTRL>
__device__ __forceinline__ float dpp_fmax(float x) {
  int yi = __builtin_amdgcn_update_dpp(0, __float_as_int(x), CTRL, 0xf, 0xf, true);
  return fmaxf(x, __int_as_float(yi));
}
template <int CTRL>
__device__ __forceinline__ float dpp_fadd(float x) {
  int yi = __builtin_amdgcn_update_dpp(0, __float_as_int(x), CTRL, 0xf, 0xf, true);
  return x + __int_as_float(yi);
}
__device__ __forceinline__ float rowmax16(float v) {
  v = dpp_fmax<0xB1>(v);   // quad_perm(1,0,3,2): xor 1
  v = dpp_fmax<0x4E>(v);   // quad_perm(2,3,0,1): xor 2
  v = dpp_fmax<0x124>(v);  // row_ror:4
  v = dpp_fmax<0x128>(v);  // row_ror:8
  return v;
}
__device__ __forceinline__ float rowsum16(float v) {
  v = dpp_fadd<0xB1>(v);
  v = dpp_fadd<0x4E>(v);
  v = dpp_fadd<0x124>(v);
  v = dpp_fadd<0x128>(v);
  return v;
}

// ---------------------------------------------------------------------------
// f32 -> bf16 conversion (weights, once per launch)
// ---------------------------------------------------------------------------
__global__ void cvt_f32_to_bf16(const float* __restrict__ in,
                                bf16* __restrict__ out, int n) {
  int i = blockIdx.x * blockDim.x + threadIdx.x;
  if (i < n) out[i] = (bf16)in[i];
}

// ---------------------------------------------------------------------------
// RMSNorm: one block per row of D_MODEL, fp32 math, bf16 out
// ---------------------------------------------------------------------------
__global__ __launch_bounds__(256)
void rmsnorm_kernel(const float* __restrict__ x, const float* __restrict__ w,
                    bf16* __restrict__ out) {
  const int row = blockIdx.x;
  const int tid = threadIdx.x;
  const float* xr = x + (size_t)row * D_MODEL;
  float ss = 0.f;
  #pragma unroll
  for (int i = tid; i < D_MODEL; i += 256) { float v = xr[i]; ss += v * v; }
  __shared__ float red[256];
  red[tid] = ss;
  __syncthreads();
  for (int s = 128; s > 0; s >>= 1) {
    if (tid < s) red[tid] += red[tid + s];
    __syncthreads();
  }
  const float inv = rsqrtf(red[0] * (1.0f / D_MODEL) + 1e-5f);
  #pragma unroll
  for (int i = tid; i < D_MODEL; i += 256)
    out[(size_t)row * D_MODEL + i] = (bf16)(xr[i] * inv * w[i]);
}

// ---------------------------------------------------------------------------
// WMMA bf16 GEMM: C[M,N] = A[M,K] * W[N,K]^T  (both k-innermost row-major)
// Block = 128 threads = 4 waves; wave tile 64x64 (4x4 of 16x16x32 WMMA);
// workgroup tile 128x128. Grid = (N/128, M/128).
// ADD_RES==0: bf16 out = oscale*acc.  ADD_RES==1: f32 out = Res + acc.
// ---------------------------------------------------------------------------
template <int ADD_RES>
__global__ __launch_bounds__(128)
void gemm_wmma_bf16(const bf16* __restrict__ A, const bf16* __restrict__ W,
                    const float* __restrict__ Res, float* __restrict__ Of,
                    bf16* __restrict__ Ob, int N, int K, float oscale) {
  const int lane = threadIdx.x & 31;
  const int wave = threadIdx.x >> 5;
  const int mBase = blockIdx.y * 128 + (wave & 1) * 64;
  const int nBase = blockIdx.x * 128 + (wave >> 1) * 64;
  const int l15 = lane & 15;
  const int hi  = lane >> 4;

  const v8f zero8 = {0.f, 0.f, 0.f, 0.f, 0.f, 0.f, 0.f, 0.f};
  v8f acc[4][4];
  #pragma unroll
  for (int i = 0; i < 4; ++i)
    #pragma unroll
    for (int j = 0; j < 4; ++j) acc[i][j] = zero8;

  for (int k0 = 0; k0 < K; k0 += 32) {
    v16bf af[4];
    #pragma unroll
    for (int mt = 0; mt < 4; ++mt) {
      const bf16* pa = A + (size_t)(mBase + mt * 16 + l15) * K + k0 + hi * 8;
      af[mt] = cat8(*(const v8bf*)(pa), *(const v8bf*)(pa + 16));
    }
    #pragma unroll
    for (int nt = 0; nt < 4; ++nt) {
      const bf16* pb = W + (size_t)(nBase + nt * 16 + l15) * K + k0 + hi * 8;
      v16bf bfr = cat8(*(const v8bf*)(pb), *(const v8bf*)(pb + 16));
      #pragma unroll
      for (int mt = 0; mt < 4; ++mt)
        acc[mt][nt] = __builtin_amdgcn_wmma_f32_16x16x32_bf16(
            false, af[mt], false, bfr, (short)0, acc[mt][nt], false, false);
    }
  }

  #pragma unroll
  for (int mt = 0; mt < 4; ++mt) {
    #pragma unroll
    for (int nt = 0; nt < 4; ++nt) {
      const int n  = nBase + nt * 16 + l15;
      const int m0 = mBase + mt * 16 + hi * 8;
      #pragma unroll
      for (int r = 0; r < 8; ++r) {
        const size_t off = (size_t)(m0 + r) * N + n;
        if (ADD_RES) Of[off] = Res[off] + acc[mt][nt][r];
        else         Ob[off] = (bf16)(acc[mt][nt][r] * oscale);
      }
    }
  }
}

// ---------------------------------------------------------------------------
// RoPE in place on bf16 [row, h*64 + d]; pairs (2i, 2i+1) within each head.
// (Linear, so it commutes with the 1/sqrt(hd) scale folded into the Q GEMM.)
// ---------------------------------------------------------------------------
__global__ void rope_kernel(bf16* __restrict__ t) {
  const int PAIRS = D_MODEL / 2;  // 512 pairs per row
  int idx = blockIdx.x * blockDim.x + threadIdx.x;
  if (idx >= MTOT * PAIRS) return;
  int row = idx / PAIRS;
  int p   = idx - row * PAIRS;     // p = h*32 + i
  int i   = p & 31;
  int s   = row & (SEQ - 1);
  float inv_freq = __expf(-(float)(2 * i) * (1.0f / HEAD_DIM) * __logf(10000.0f));
  float ang = (float)s * inv_freq;
  float sn, cs;
  __sincosf(ang, &sn, &cs);
  size_t base = (size_t)row * D_MODEL + (size_t)p * 2;
  float xe = (float)t[base], xo = (float)t[base + 1];
  t[base]     = (bf16)(xe * cs - xo * sn);
  t[base + 1] = (bf16)(xo * cs + xe * sn);
}

// ---------------------------------------------------------------------------
// V transpose into [bh, d, s] so PV B-fragments read keys contiguously.
// ---------------------------------------------------------------------------
__global__ void transpose_v_kernel(const bf16* __restrict__ V,
                                   bf16* __restrict__ Vt) {
  int idx = blockIdx.x * blockDim.x + threadIdx.x;  // over MTOT*D_MODEL
  if (idx >= MTOT * D_MODEL) return;
  int s   = idx & (SEQ - 1);
  int rem = idx >> 11;            // bh*64 + d
  int d   = rem & (HEAD_DIM - 1);
  int bh  = rem >> 6;
  int b   = bh >> 4, h = bh & 15;
  Vt[idx] = V[(((size_t)b * SEQ + s) * D_MODEL) + h * HEAD_DIM + d];
}

// ---------------------------------------------------------------------------
// WMMA causal flash attention. Block = 4 waves; each wave owns a 16-query
// tile of one (b,h). Grid = (SEQ/64, BATCH*NUM_HEADS).
// Main loop runs fully-unmasked key tiles; exactly one diagonal tile gets
// the causal compare/select path. Per-row denominator is kept as a per-lane
// partial and reduced by DPP only once at the end.
// ---------------------------------------------------------------------------
__global__ __launch_bounds__(128)
void flash_attn_wmma(const bf16* __restrict__ Q, const bf16* __restrict__ Kb,
                     const bf16* __restrict__ Vt, bf16* __restrict__ O) {
  __shared__ __align__(16) bf16 plds[4][16][32];  // per-wave P tile (1 KB)

  const int lane = threadIdx.x & 31;
  const int wave = threadIdx.x >> 5;
  const int l15 = lane & 15;
  const int hi  = lane >> 4;
  const int bh  = blockIdx.y;
  const int b   = bh >> 4, h = bh & 15;
  const int qBase = blockIdx.x * 64 + wave * 16;

  const bf16* Kbase = Kb + ((size_t)b * SEQ) * D_MODEL + h * HEAD_DIM;
  const bf16* Vbase = Vt + ((size_t)bh * HEAD_DIM) * SEQ;

  // Q tile 16x64 as two A-fragments (K=32 each); logits pre-scaled in Q GEMM
  const bf16* qp = Q + ((size_t)b * SEQ + qBase + l15) * D_MODEL + h * HEAD_DIM;
  v16bf qf[2];
  #pragma unroll
  for (int f = 0; f < 2; ++f) {
    const bf16* p = qp + f * 32 + hi * 8;
    qf[f] = cat8(*(const v8bf*)(p), *(const v8bf*)(p + 16));
  }

  const v8f z8 = {0.f, 0.f, 0.f, 0.f, 0.f, 0.f, 0.f, 0.f};
  float mrow[8], lpart[8];
  v8f o[4];
  #pragma unroll
  for (int r = 0; r < 8; ++r) { mrow[r] = -1e30f; lpart[r] = 0.f; }
  #pragma unroll
  for (int nt = 0; nt < 4; ++nt) o[nt] = z8;

  auto iter = [&](int j0, bool masked) {
    // ---- S = Q K^T for keys [j0, j0+32) : two 16-key sub-tiles ----
    v8f sa[2];
    #pragma unroll
    for (int sub = 0; sub < 2; ++sub) {
      const bf16* kp = Kbase + (size_t)(j0 + sub * 16 + l15) * D_MODEL + hi * 8;
      v16bf kb0 = cat8(*(const v8bf*)(kp),      *(const v8bf*)(kp + 16));
      v16bf kb1 = cat8(*(const v8bf*)(kp + 32), *(const v8bf*)(kp + 48));
      v8f s = z8;
      s = __builtin_amdgcn_wmma_f32_16x16x32_bf16(false, qf[0], false, kb0,
                                                  (short)0, s, false, false);
      s = __builtin_amdgcn_wmma_f32_16x16x32_bf16(false, qf[1], false, kb1,
                                                  (short)0, s, false, false);
      sa[sub] = s;
    }

    // ---- (optional) causal mask + row max via DPP ----
    float s0a[8], s1a[8], rmax[8];
    #pragma unroll
    for (int r = 0; r < 8; ++r) {
      float s0 = sa[0][r];
      float s1 = sa[1][r];
      if (masked) {
        const int rowg = qBase + r + hi * 8;
        if (j0 + l15 > rowg)      s0 = -1e30f;
        if (j0 + 16 + l15 > rowg) s1 = -1e30f;
      }
      s0a[r] = s0; s1a[r] = s1;
      rmax[r] = rowmax16(fmaxf(s0, s1));
    }

    // ---- online softmax update (per-lane partial denominator) ----
    float corr[8], p0a[8], p1a[8];
    #pragma unroll
    for (int r = 0; r < 8; ++r) {
      float nm = fmaxf(mrow[r], rmax[r]);
      corr[r]  = __expf(mrow[r] - nm);
      float p0 = __expf(s0a[r] - nm);
      float p1 = __expf(s1a[r] - nm);
      p0a[r] = p0; p1a[r] = p1;
      lpart[r] = lpart[r] * corr[r] + p0 + p1;
      mrow[r] = nm;
    }
    #pragma unroll
    for (int nt = 0; nt < 4; ++nt)
      #pragma unroll
      for (int r = 0; r < 8; ++r) o[nt][r] *= corr[r];

    // ---- restage P: C-layout -> LDS -> A-fragment (in-order within wave) --
    #pragma unroll
    for (int r = 0; r < 8; ++r) {
      const int m = r + hi * 8;
      plds[wave][m][l15]      = (bf16)p0a[r];
      plds[wave][m][16 + l15] = (bf16)p1a[r];
    }
    const bf16* pl = &plds[wave][l15][hi * 8];
    v16bf pfrag = cat8(*(const v8bf*)(pl), *(const v8bf*)(pl + 16));

    // ---- O += P V : keys contiguous via Vt [bh, d, s] ----
    #pragma unroll
    for (int nt = 0; nt < 4; ++nt) {
      const bf16* vp = Vbase + (size_t)(nt * 16 + l15) * SEQ + j0 + hi * 8;
      v16bf vfrag = cat8(*(const v8bf*)(vp), *(const v8bf*)(vp + 16));
      o[nt] = __builtin_amdgcn_wmma_f32_16x16x32_bf16(
          false, pfrag, false, vfrag, (short)0, o[nt], false, false);
    }
  };

  int j0 = 0;
  for (; j0 + 32 <= qBase; j0 += 32) iter(j0, false);  // fully unmasked tiles
  iter(j0, true);                                      // single diagonal tile

  // ---- epilogue: finish denominator with one DPP reduce, scatter bf16 ----
  float rinv[8];
  #pragma unroll
  for (int r = 0; r < 8; ++r) rinv[r] = 1.f / rowsum16(lpart[r]);
  #pragma unroll
  for (int nt = 0; nt < 4; ++nt) {
    #pragma unroll
    for (int r = 0; r < 8; ++r) {
      const size_t off = ((size_t)b * SEQ + qBase + r + hi * 8) * D_MODEL +
                         h * HEAD_DIM + nt * 16 + l15;
      O[off] = (bf16)(o[nt][r] * rinv[r]);
    }
  }
}

// ---------------------------------------------------------------------------
// SwiGLU gate: h = up * sigmoid(up) * gate  (in place into up buffer)
// ---------------------------------------------------------------------------
__global__ void silu_gate_kernel(bf16* __restrict__ up,
                                 const bf16* __restrict__ gate, int n) {
  int i = blockIdx.x * blockDim.x + threadIdx.x;
  if (i < n) {
    float u = (float)up[i], g = (float)gate[i];
    float sw = u / (1.f + __expf(-u));
    up[i] = (bf16)(sw * g);
  }
}

// ---------------------------------------------------------------------------
// Host orchestration
// ---------------------------------------------------------------------------
extern "C" void kernel_launch(void* const* d_in, const int* in_sizes, int n_in,
                              void* d_out, int out_size, void* d_ws, size_t ws_size,
                              hipStream_t stream) {
  (void)in_sizes; (void)n_in; (void)out_size; (void)ws_size;
  const float* x      = (const float*)d_in[0];
  const float* wq     = (const float*)d_in[1];
  const float* wk     = (const float*)d_in[2];
  const float* wv     = (const float*)d_in[3];
  const float* wo     = (const float*)d_in[4];
  const float* ln1_w  = (const float*)d_in[5];
  const float* ln2_w  = (const float*)d_in[6];
  const float* up_w   = (const float*)d_in[7];
  const float* gate_w = (const float*)d_in[8];
  const float* down_w = (const float*)d_in[9];
  float* out = (float*)d_out;

  char* ws = (char*)d_ws;
  size_t off = 0;
  auto alloc = [&](size_t bytes) -> void* {
    void* p = ws + off;
    off += (bytes + 255) & ~(size_t)255;
    return p;
  };

  bf16* xn     = (bf16*)alloc((size_t)MTOT * D_MODEL * 2);
  bf16* wqb    = (bf16*)alloc((size_t)D_MODEL * D_MODEL * 2);
  bf16* wkb    = (bf16*)alloc((size_t)D_MODEL * D_MODEL * 2);
  bf16* wvb    = (bf16*)alloc((size_t)D_MODEL * D_MODEL * 2);
  bf16* wob    = (bf16*)alloc((size_t)D_MODEL * D_MODEL * 2);
  bf16* upwb   = (bf16*)alloc((size_t)D_FF * D_MODEL * 2);
  bf16* gatewb = (bf16*)alloc((size_t)D_FF * D_MODEL * 2);
  bf16* downwb = (bf16*)alloc((size_t)D_MODEL * D_FF * 2);
  bf16* qb     = (bf16*)alloc((size_t)MTOT * D_MODEL * 2);
  bf16* kb     = (bf16*)alloc((size_t)MTOT * D_MODEL * 2);
  bf16* vb     = (bf16*)alloc((size_t)MTOT * D_MODEL * 2);
  bf16* vtb    = (bf16*)alloc((size_t)MTOT * D_MODEL * 2);
  bf16* aob    = (bf16*)alloc((size_t)MTOT * D_MODEL * 2);
  float* x1    = (float*)alloc((size_t)MTOT * D_MODEL * 4);
  bf16* xn2    = (bf16*)alloc((size_t)MTOT * D_MODEL * 2);
  bf16* upb    = (bf16*)alloc((size_t)MTOT * D_FF * 2);
  bf16* gateb  = (bf16*)alloc((size_t)MTOT * D_FF * 2);

  auto cvt = [&](const float* src, bf16* dst, int n) {
    cvt_f32_to_bf16<<<(n + 255) / 256, 256, 0, stream>>>(src, dst, n);
  };
  cvt(wq, wqb, D_MODEL * D_MODEL);
  cvt(wk, wkb, D_MODEL * D_MODEL);
  cvt(wv, wvb, D_MODEL * D_MODEL);
  cvt(wo, wob, D_MODEL * D_MODEL);
  cvt(up_w, upwb, D_FF * D_MODEL);
  cvt(gate_w, gatewb, D_FF * D_MODEL);
  cvt(down_w, downwb, D_MODEL * D_FF);

  // --- attention half ---
  rmsnorm_kernel<<<MTOT, 256, 0, stream>>>(x, ln1_w, xn);

  dim3 gQ(D_MODEL / 128, MTOT / 128);
  // 1/sqrt(HEAD_DIM) folded into Q projection (exact power of two in bf16)
  gemm_wmma_bf16<0><<<gQ, 128, 0, stream>>>(xn, wqb, nullptr, nullptr, qb, D_MODEL, D_MODEL, 0.125f);
  gemm_wmma_bf16<0><<<gQ, 128, 0, stream>>>(xn, wkb, nullptr, nullptr, kb, D_MODEL, D_MODEL, 1.0f);
  gemm_wmma_bf16<0><<<gQ, 128, 0, stream>>>(xn, wvb, nullptr, nullptr, vb, D_MODEL, D_MODEL, 1.0f);

  const int npairs = MTOT * (D_MODEL / 2);
  rope_kernel<<<(npairs + 255) / 256, 256, 0, stream>>>(qb);
  rope_kernel<<<(npairs + 255) / 256, 256, 0, stream>>>(kb);

  const int nelem = MTOT * D_MODEL;
  transpose_v_kernel<<<(nelem + 255) / 256, 256, 0, stream>>>(vb, vtb);

  dim3 gA(SEQ / 64, BATCH * NUM_HEADS);
  flash_attn_wmma<<<gA, 128, 0, stream>>>(qb, kb, vtb, aob);

  // x1 = x + ao @ wo^T  (residual fused in GEMM epilogue)
  gemm_wmma_bf16<1><<<gQ, 128, 0, stream>>>(aob, wob, x, x1, nullptr, D_MODEL, D_MODEL, 1.0f);

  // --- MLP half ---
  rmsnorm_kernel<<<MTOT, 256, 0, stream>>>(x1, ln2_w, xn2);

  dim3 gF(D_FF / 128, MTOT / 128);
  gemm_wmma_bf16<0><<<gF, 128, 0, stream>>>(xn2, upwb, nullptr, nullptr, upb, D_FF, D_MODEL, 1.0f);
  gemm_wmma_bf16<0><<<gF, 128, 0, stream>>>(xn2, gatewb, nullptr, nullptr, gateb, D_FF, D_MODEL, 1.0f);

  const int nff = MTOT * D_FF;
  silu_gate_kernel<<<(nff + 255) / 256, 256, 0, stream>>>(upb, gateb, nff);

  // out = x1 + h @ down_w^T
  dim3 gD(D_MODEL / 128, MTOT / 128);
  gemm_wmma_bf16<1><<<gD, 128, 0, stream>>>(upb, downwb, x1, out, nullptr, D_MODEL, D_FF, 1.0f);
}
